// SimpleODELSTM_28561532518554
// MI455X (gfx1250) — matrix operand order repaired and checked by reading, not compiled, not run on hardware
//
#include <hip/hip_runtime.h>

typedef __attribute__((ext_vector_type(16))) _Float16 v16h;
typedef __attribute__((ext_vector_type(8)))  _Float16 v8h;
typedef __attribute__((ext_vector_type(8)))  float    v8f;

#define HID    64
#define KAUG   96      // 64 (h) + x0 + x1 + bias-one + 29 zero pad
#define TSTEPS 512
#define BTOT   4096
#define WPB    4       // waves per block

#if __has_builtin(__builtin_amdgcn_tanhf)
#define TANHF(v) __builtin_amdgcn_tanhf(v)
#elif __has_builtin(__builtin_amdgcn_tanh_f32)
#define TANHF(v) __builtin_amdgcn_tanh_f32(v)
#else
#define TANHF(v) (2.0f / (1.0f + __expf(-2.0f * (v))) - 1.0f)
#endif

__device__ __forceinline__ float fast_sig(float x) {
    // sigmoid(x) == 0.5 * tanh(x/2) + 0.5
    return fmaf(0.5f, TANHF(0.5f * x), 0.5f);
}

__device__ __forceinline__ v8f splat8(float v) {
    v8f a;
#pragma unroll
    for (int i = 0; i < 8; ++i) a[i] = v;
    return a;
}

// B fragment (K=32 x N=16 slice) of a row-major [N][stride] f16 matrix in LDS.
__device__ __forceinline__ v16h ld_bfrag(const _Float16* base, int stride, int n, int half, int kstep) {
    const _Float16* p = base + n * stride + kstep * 32 + half * 16;
    v8h lo = *(const v8h*)p;
    v8h hi = *(const v8h*)(p + 8);
    return __builtin_shufflevector(lo, hi, 0,1,2,3,4,5,6,7,8,9,10,11,12,13,14,15);
}

// A fragment (M=16 x K=32 slice) from row-major [16][KAUG] f16 staging tile.
__device__ __forceinline__ v16h ld_afrag(const _Float16* stag, int m, int half, int kstep) {
    const _Float16* p = stag + m * KAUG + kstep * 32 + half * 16;
    v8h lo = *(const v8h*)p;        // K base .. base+7
    v8h hi = *(const v8h*)(p + 8);  // K base+8 .. base+15
    return __builtin_shufflevector(lo, hi, 0,1,2,3,4,5,6,7,8,9,10,11,12,13,14,15);
}

__device__ __forceinline__ v8f wmma16(v16h a, v16h b, v8f c) {
    return __builtin_amdgcn_wmma_f32_16x16x32_f16(false, a, false, b, (short)0, c, false, false);
}

__global__ __launch_bounds__(32 * WPB)
void odelstm_kernel(const float* __restrict__ x,
                    const float* __restrict__ W_ih, const float* __restrict__ W_hh,
                    const float* __restrict__ b_ih, const float* __restrict__ b_hh,
                    const float* __restrict__ W1,   const float* __restrict__ b1,
                    const float* __restrict__ W2,   const float* __restrict__ b2,
                    const float* __restrict__ Wf,   const float* __restrict__ bf,
                    float* __restrict__ out)
{
    __shared__ __align__(16) _Float16 sWhhA[4 * HID * KAUG];   // 48 KB augmented [256][96]
    __shared__ __align__(16) _Float16 sW1[HID * HID];          // 8 KB
    __shared__ __align__(16) _Float16 sW2[HID * HID];          // 8 KB
    __shared__ __align__(16) _Float16 sWf[HID * HID];          // 8 KB
    __shared__ __align__(16) _Float16 sStag[WPB][16 * KAUG];   // 3 KB per wave

    const int tid = threadIdx.x;

    // Augmented gate weights: row n = [ W_hh[n][0..63] | W_ih[n][0] | W_ih[n][1] |
    //                                   b_ih[n]+b_hh[n] | zeros ]
    for (int i = tid; i < 4 * HID * KAUG; i += blockDim.x) {
        int n = i / KAUG, k = i % KAUG;
        float v;
        if (k < HID)          v = W_hh[n * HID + k];
        else if (k == HID)    v = W_ih[n * 2 + 0];
        else if (k == HID+1)  v = W_ih[n * 2 + 1];
        else if (k == HID+2)  v = b_ih[n] + b_hh[n];
        else                  v = 0.0f;
        sWhhA[i] = (_Float16)v;
    }
    for (int i = tid; i < HID * HID; i += blockDim.x) {
        sW1[i] = (_Float16)W1[i];
        sW2[i] = (_Float16)W2[i];
        sWf[i] = (_Float16)Wf[i];
    }
    __syncthreads();

    const int wave = tid >> 5;
    const int lane = tid & 31;
    const int half = lane >> 4;   // 0: lanes 0-15, 1: lanes 16-31
    const int l16  = lane & 15;
    _Float16* stag = sStag[wave];

    const int tile = blockIdx.x * WPB + wave;
    const int b0   = tile * 16;

    // One-time constant columns of the augmented A tile: col 66 = 1.0, 67..95 = 0.
    stag[l16 * KAUG + HID + 2] = (_Float16)1.0f;
    for (int k = HID + 3; k < KAUG; ++k) stag[l16 * KAUG + k] = (_Float16)0.0f;

    float b1_l[4], b2_l[4], bf_l[4];
#pragma unroll
    for (int j = 0; j < 4; ++j) {
        int n = j * 16 + l16;
        b1_l[j] = b1[n];
        b2_l[j] = b2[n];
        bf_l[j] = bf[n];
    }

    v8f cS[4], hC[4];
#pragma unroll
    for (int j = 0; j < 4; ++j) { cS[j] = splat8(0.0f); hC[j] = splat8(0.0f); }
    v16h hA0, hA1;
#pragma unroll
    for (int i = 0; i < 16; ++i) { hA0[i] = (_Float16)0.0f; hA1[i] = (_Float16)0.0f; }

    const float dt = 1.0f / 3.0f;
    const float* xb = x + (size_t)b0 * (2 * TSTEPS);
    const size_t xoff = (size_t)l16 * (2 * TSTEPS) + (size_t)half * TSTEPS;
    float xv = xb[xoff];   // x for t=0, prefetched

    auto store_tile = [&](int j, const v8f& v) {
#pragma unroll
        for (int r = 0; r < 8; ++r)
            stag[(r + 8 * half) * KAUG + j * 16 + l16] = (_Float16)v[r];
    };

    // Two output tiles from one A pair; all 4 B fragments hoisted, chains interleaved.
    auto mmpair = [&](const _Float16* W, int j0, int j1, v16h A0, v16h A1,
                      float bias0, float bias1, v8f& o0, v8f& o1) {
        v16h w00 = ld_bfrag(W, HID, j0 * 16 + l16, half, 0);
        v16h w01 = ld_bfrag(W, HID, j0 * 16 + l16, half, 1);
        v16h w10 = ld_bfrag(W, HID, j1 * 16 + l16, half, 0);
        v16h w11 = ld_bfrag(W, HID, j1 * 16 + l16, half, 1);
        o0 = wmma16(A0, w00, splat8(bias0));
        o1 = wmma16(A0, w10, splat8(bias1));
        o0 = wmma16(A1, w01, o0);
        o1 = wmma16(A1, w11, o1);
    };

    for (int t = 0; t < TSTEPS; ++t) {
        // Commit this step's x into the A tile, then prefetch next step's x.
        stag[l16 * KAUG + HID + half] = (_Float16)xv;
        {
            int tn = (t + 1 < TSTEPS) ? (t + 1) : t;
            xv = xb[xoff + tn];
        }
        v16h hA2 = ld_afrag(stag, l16, half, 2);   // K = 64..95 (x, 1, pad)

        // Two gate fragments at once: 6 B frags hoisted, 2 interleaved WMMA chains.
        auto gate2 = [&](int ga, int gb, v8f& accA, v8f& accB) {
            int na = ga * 16 + l16, nb = gb * 16 + l16;
            v16h a0 = ld_bfrag(sWhhA, KAUG, na, half, 0);
            v16h a1 = ld_bfrag(sWhhA, KAUG, na, half, 1);
            v16h a2 = ld_bfrag(sWhhA, KAUG, na, half, 2);
            v16h b0 = ld_bfrag(sWhhA, KAUG, nb, half, 0);
            v16h b1f = ld_bfrag(sWhhA, KAUG, nb, half, 1);
            v16h b2f = ld_bfrag(sWhhA, KAUG, nb, half, 2);
            accA = wmma16(hA0, a0, splat8(0.0f));
            accB = wmma16(hA0, b0, splat8(0.0f));
            accA = wmma16(hA1, a1, accA);
            accB = wmma16(hA1, b1f, accB);
            accA = wmma16(hA2, a2, accA);
            accB = wmma16(hA2, b2f, accB);
        };

        // ---- LSTM cell, per hidden-column group of 16 ----
#pragma unroll
        for (int j = 0; j < 4; ++j) {
            v8f gi, gg;
            gate2(j, 8 + j, gi, gg);     // i and g gates
            v8f ig;
#pragma unroll
            for (int r = 0; r < 8; ++r) ig[r] = fast_sig(gi[r]) * TANHF(gg[r]);
            v8f gf, go;
            gate2(4 + j, 12 + j, gf, go); // f and o gates
            v8f cj = cS[j], hj;
#pragma unroll
            for (int r = 0; r < 8; ++r) {
                float cv = fmaf(fast_sig(gf[r]), cj[r], ig[r]);
                cj[r] = cv;
                hj[r] = fast_sig(go[r]) * TANHF(cv);
            }
            cS[j] = cj;
            hC[j] = hj;
            store_tile(j, hj);
        }
        hA0 = ld_afrag(stag, l16, half, 0);
        hA1 = ld_afrag(stag, l16, half, 1);

        // ---- ODE: 3 fixed Euler steps: h += dt * (tanh(h W1^T + b1) W2^T + b2) ----
#pragma unroll
        for (int s = 0; s < 3; ++s) {
#pragma unroll
            for (int jp = 0; jp < 2; ++jp) {
                const int j0 = 2 * jp, j1 = 2 * jp + 1;
                v8f z0, z1;
                mmpair(sW1, j0, j1, hA0, hA1, b1_l[j0], b1_l[j1], z0, z1);
#pragma unroll
                for (int r = 0; r < 8; ++r) { z0[r] = TANHF(z0[r]); z1[r] = TANHF(z1[r]); }
                store_tile(j0, z0);
                store_tile(j1, z1);
            }
            v16h uA0 = ld_afrag(stag, l16, half, 0);
            v16h uA1 = ld_afrag(stag, l16, half, 1);
#pragma unroll
            for (int jp = 0; jp < 2; ++jp) {
                const int j0 = 2 * jp, j1 = 2 * jp + 1;
                v8f v0, v1;
                mmpair(sW2, j0, j1, uA0, uA1, b2_l[j0], b2_l[j1], v0, v1);
#pragma unroll
                for (int r = 0; r < 8; ++r) {
                    hC[j0][r] = fmaf(dt, v0[r], hC[j0][r]);
                    hC[j1][r] = fmaf(dt, v1[r], hC[j1][r]);
                }
                store_tile(j0, hC[j0]);
                store_tile(j1, hC[j1]);
            }
            hA0 = ld_afrag(stag, l16, half, 0);
            hA1 = ld_afrag(stag, l16, half, 1);
        }

        // ---- fc: out[b, t, :] = h Wf^T + bf ----
#pragma unroll
        for (int jp = 0; jp < 2; ++jp) {
            const int j0 = 2 * jp, j1 = 2 * jp + 1;
            v8f o0, o1;
            mmpair(sWf, j0, j1, hA0, hA1, bf_l[j0], bf_l[j1], o0, o1);
#pragma unroll
            for (int r = 0; r < 8; ++r) {
                int m = r + 8 * half;
                size_t base = (((size_t)(b0 + m)) * TSTEPS + t) * HID + l16;
                out[base + j0 * 16] = o0[r];
                out[base + j1 * 16] = o1[r];
            }
        }
    }
}

extern "C" void kernel_launch(void* const* d_in, const int* in_sizes, int n_in,
                              void* d_out, int out_size, void* d_ws, size_t ws_size,
                              hipStream_t stream) {
    const float* x    = (const float*)d_in[0];
    const float* W_ih = (const float*)d_in[1];
    const float* W_hh = (const float*)d_in[2];
    const float* b_ih = (const float*)d_in[3];
    const float* b_hh = (const float*)d_in[4];
    const float* W1   = (const float*)d_in[5];
    const float* b1   = (const float*)d_in[6];
    const float* W2   = (const float*)d_in[7];
    const float* b2   = (const float*)d_in[8];
    const float* Wf   = (const float*)d_in[9];
    const float* bf   = (const float*)d_in[10];
    float* out = (float*)d_out;

    dim3 grid(BTOT / (16 * WPB));   // 64 workgroups
    dim3 block(32 * WPB);           // 4 waves (wave32)
    hipLaunchKernelGGL(odelstm_kernel, grid, block, 0, stream,
                       x, W_ih, W_hh, b_ih, b_hh, W1, b1, W2, b2, Wf, bf, out);
}
